// ReformerClassifier_1881195676095
// MI455X (gfx1250) — matrix operand order repaired and checked by reading, not compiled, not run on hardware
//
#include <hip/hip_runtime.h>
#include <hip/hip_bf16.h>
#include <cstdint>

// ---------------- problem constants ----------------
#define VOCAB 32000
#define NCLS  50
#define Bz    4
#define Sz    4096
#define Dz    512
#define Hh    8
#define DHz   64          // D / H
#define Lz    2
#define BUCKET 64
#define NHASH 4
#define NBz   64          // S / BUCKET
#define NCHUNK (NHASH * NBz)   // 256
#define BHz   (Bz * Hh)        // 32
#define BSz   (Bz * Sz)        // 16384
#define SELF_MASK (-5e4f)
#define MASK_VAL  (-1e9f)

typedef __attribute__((ext_vector_type(16))) _Float16 v16h;
typedef __attribute__((ext_vector_type(8)))  _Float16 v8h;
typedef __attribute__((ext_vector_type(8)))  float    v8f;

static __device__ __forceinline__ v16h cat16(v8h lo, v8h hi) {
  return __builtin_shufflevector(lo, hi, 0, 1, 2, 3, 4, 5, 6, 7,
                                 8, 9, 10, 11, 12, 13, 14, 15);
}

// ---------------- Tensor Data Mover (gfx1250) ----------------
#if defined(__has_builtin)
#if __has_builtin(__builtin_amdgcn_tensor_load_to_lds) && \
    __has_builtin(__builtin_amdgcn_s_wait_tensorcnt)
#define HAVE_TDM 1
#endif
#endif

#ifdef HAVE_TDM
typedef __attribute__((ext_vector_type(4))) unsigned int u32x4;
typedef __attribute__((ext_vector_type(8))) int          i32x8;
typedef __attribute__((ext_vector_type(4))) int          i32x4;

// Load a 2D tile (rows x 32 f16 elements, row stride strideElems) from
// global memory into LDS at byte offset lds_off, via the TDM (D# descriptor
// per CDNA5 ISA sec 8.3-8.4). Tracked with TENSORcnt.
static __device__ __forceinline__ void tdm_load_2d(const _Float16* g,
                                                   unsigned lds_off,
                                                   unsigned rows,
                                                   unsigned strideElems) {
  unsigned long long ga = (unsigned long long)(uintptr_t)g;
  u32x4 g0;
  g0[0] = 1u;                                    // count=1, user mode
  g0[1] = lds_off;                               // lds_addr (bytes)
  g0[2] = (unsigned)ga;                          // global_addr[31:0]
  g0[3] = (unsigned)((ga >> 32) & 0x01FFFFFFu) | (2u << 30);  // [56:32]+type=2
  unsigned td0 = 32u, td1 = rows, t0 = 32u, t1 = rows;
  unsigned long long st0 = (unsigned long long)strideElems;
  i32x8 g1;
  g1[0] = (int)(1u << 16);                       // data_size=1 -> 2 bytes
  g1[1] = (int)((td0 & 0xFFFFu) << 16);          // tensor_dim0 lo
  g1[2] = (int)(((td0 >> 16) & 0xFFFFu) | ((td1 & 0xFFFFu) << 16));
  g1[3] = (int)(((td1 >> 16) & 0xFFFFu) | ((t0 & 0xFFFFu) << 16));
  g1[4] = (int)(t1 & 0xFFFFu);                   // tile_dim1 (tile_dim2=0)
  g1[5] = (int)(st0 & 0xFFFFFFFFu);              // tensor_dim0_stride lo
  g1[6] = (int)((st0 >> 32) & 0xFFFFu);          // stride hi (dim1_stride=0)
  g1[7] = 0;
  i32x4 g2 = {0, 0, 0, 0};
  i32x4 g3 = {0, 0, 0, 0};
#if defined(__clang_major__) && __clang_major__ >= 23
  i32x8 g4 = {0, 0, 0, 0, 0, 0, 0, 0};
  __builtin_amdgcn_tensor_load_to_lds(g0, g1, g2, g3, g4, 0);
#else
  __builtin_amdgcn_tensor_load_to_lds(g0, g1, g2, g3, 0);
#endif
}
#endif  // HAVE_TDM

// ============================================================
// Embedding + sinusoidal positional encoding -> X1 and X2
// ============================================================
__global__ void embed_kernel(const int* __restrict__ src,
                             const float* __restrict__ emb,
                             float* __restrict__ x1, float* __restrict__ x2) {
  int tokpos = blockIdx.x;              // 0..BS-1
  int s = tokpos % Sz;
  int tok = src[tokpos];
  const float* erow = emb + (size_t)tok * Dz;
  for (int it = 0; it < Dz / 256; ++it) {
    int d = threadIdx.x + it * 256;
    int de = d & ~1;
    float f = __expf(-(float)de * (9.210340372f / (float)Dz));  // ln(10000)
    float ang = (float)s * f;
    float pe = (d & 1) ? __cosf(ang) : __sinf(ang);
    float v = erow[d] + pe;
    x1[(size_t)tokpos * Dz + d] = v;
    x2[(size_t)tokpos * Dz + d] = v;
  }
}

// ============================================================
// LayerNorm over D=512, one block (256 thr) per token; f16 out
// ============================================================
__global__ void layernorm_kernel(const float* __restrict__ x,
                                 const float* __restrict__ g,
                                 const float* __restrict__ b,
                                 _Float16* __restrict__ out) {
  __shared__ float red[256];
  int tokpos = blockIdx.x;
  const float* row = x + (size_t)tokpos * Dz;
  float v0 = row[threadIdx.x];
  float v1 = row[threadIdx.x + 256];
  red[threadIdx.x] = v0 + v1;
  __syncthreads();
  for (int off = 128; off > 0; off >>= 1) {
    if (threadIdx.x < off) red[threadIdx.x] += red[threadIdx.x + off];
    __syncthreads();
  }
  float mean = red[0] * (1.0f / Dz);
  __syncthreads();
  float d0 = v0 - mean, d1 = v1 - mean;
  red[threadIdx.x] = d0 * d0 + d1 * d1;
  __syncthreads();
  for (int off = 128; off > 0; off >>= 1) {
    if (threadIdx.x < off) red[threadIdx.x] += red[threadIdx.x + off];
    __syncthreads();
  }
  float rstd = rsqrtf(red[0] * (1.0f / Dz) + 1e-5f);
  _Float16* orow = out + (size_t)tokpos * Dz;
  int d = threadIdx.x;
  orow[d]       = (_Float16)(d0 * rstd * g[d] + b[d]);
  orow[d + 256] = (_Float16)(d1 * rstd * g[d + 256] + b[d + 256]);
}

// ============================================================
// Weight pack: W[K,N] f32 -> WT[N,K] f16
// ============================================================
__global__ void packw_kernel(const float* __restrict__ W,
                             _Float16* __restrict__ WT, int K, int N) {
  int idx = blockIdx.x * blockDim.x + threadIdx.x;
  if (idx >= K * N) return;
  int n = idx / K, k = idx % K;
  WT[(size_t)n * K + k] = (_Float16)W[(size_t)k * N + n];
}

// ============================================================
// WMMA GEMM: C[M,N] = act(A[M,K] @ WT[N,K]^T + bias) (+resid)
// block = 4 waves computing 128x64; per K-step the 128x32 A slab
// and 64x32 WT tile are staged in LDS by the TDM (wave 0 issues,
// s_wait_tensorcnt, workgroup barrier), then each wave runs
// 8 v_wmma per K-step from ds_load_b128 fragments.
// act: 0=none, 1=exact GELU. Writes Cf (f32) and/or Ch (f16).
// ============================================================
__global__ void gemm_wmma_kernel(const _Float16* __restrict__ A,
                                 const _Float16* __restrict__ WT,
                                 const float* __restrict__ bias,
                                 const float* __restrict__ resid,
                                 float* __restrict__ Cf,
                                 _Float16* __restrict__ Ch,
                                 int M, int N, int K, int act) {
  __shared__ _Float16 As[128 * 32];   // 8 KB A slab
  __shared__ _Float16 Bs[64 * 32];    // 4 KB WT tile
  int n0 = blockIdx.x * 64;
  int m0 = blockIdx.y * 128;
  int tid = threadIdx.x;
  int wave = tid >> 5;
  int lane = tid & 31;
  int lm = lane & 15;
  bool hi = lane >= 16;
  int aB = hi ? 8 : 0;
  int bB = hi ? 8 : 0;

  v8f acc[2][4] = {};
#ifdef HAVE_TDM
  unsigned asOff = (unsigned)(uintptr_t)(void*)As;
  unsigned bsOff = (unsigned)(uintptr_t)(void*)Bs;
#endif

  for (int k0 = 0; k0 < K; k0 += 32) {
    __syncthreads();   // previous K-step fully consumed before overwrite
    __builtin_prefetch(A + (size_t)m0 * K + k0 + 64, 0, 1);  // global_prefetch_b8
#ifdef HAVE_TDM
    if (wave == 0) {
      tdm_load_2d(A + (size_t)m0 * K + k0, asOff, 128u, (unsigned)K);
      tdm_load_2d(WT + (size_t)n0 * K + k0, bsOff, 64u, (unsigned)K);
      __builtin_amdgcn_s_wait_tensorcnt((short)0);
    }
#else
    // cooperative staging fallback: 16-byte chunks
    for (int c = tid; c < 512; c += 128) {           // A: 128 rows x 4 chunks
      int row = c >> 2, q = (c & 3) * 8;
      *(v8h*)(As + row * 32 + q) =
          *(const v8h*)(A + (size_t)(m0 + row) * K + k0 + q);
    }
    for (int c = tid; c < 256; c += 128) {           // B: 64 rows x 4 chunks
      int row = c >> 2, q = (c & 3) * 8;
      *(v8h*)(Bs + row * 32 + q) =
          *(const v8h*)(WT + (size_t)(n0 + row) * K + k0 + q);
    }
#endif
    __syncthreads();

    v16h aF[2], bF[4];
#pragma unroll
    for (int mt = 0; mt < 2; ++mt) {
      const _Float16* p = As + (wave * 32 + mt * 16 + lm) * 32 + aB;
      aF[mt] = cat16(*(const v8h*)p, *(const v8h*)(p + 16));
    }
#pragma unroll
    for (int nt = 0; nt < 4; ++nt) {
      const _Float16* q = Bs + (nt * 16 + lm) * 32 + bB;
      bF[nt] = cat16(*(const v8h*)q, *(const v8h*)(q + 16));
    }
#pragma unroll
    for (int mt = 0; mt < 2; ++mt)
#pragma unroll
      for (int nt = 0; nt < 4; ++nt)
        acc[mt][nt] = __builtin_amdgcn_wmma_f32_16x16x32_f16(
            false, aF[mt], false, bF[nt], (short)0, acc[mt][nt], false, false);
  }

#pragma unroll
  for (int mt = 0; mt < 2; ++mt)
#pragma unroll
    for (int nt = 0; nt < 4; ++nt)
#pragma unroll
      for (int r = 0; r < 8; ++r) {
        int row = m0 + wave * 32 + mt * 16 + r + (hi ? 8 : 0);
        int col = n0 + nt * 16 + lm;
        float v = acc[mt][nt][r];
        if (bias) v += bias[col];
        if (act == 1) v = 0.5f * v * (1.0f + erff(v * 0.7071067811865475f));
        if (resid) v += resid[(size_t)row * N + col];
        if (Cf) Cf[(size_t)row * N + col] = v;
        if (Ch) Ch[(size_t)row * N + col] = (_Float16)v;
      }
}

// ============================================================
// LSH hashing: bucket = argmax over [r, -r] of rotations
// ============================================================
__global__ void hash_kernel(const float* __restrict__ QK,
                            const float* __restrict__ rot,   // [DH, NHASH, NB/2]
                            int* __restrict__ buckets) {
  int idx = blockIdx.x * blockDim.x + threadIdx.x;
  if (idx >= BHz * NHASH * Sz) return;
  int bh = idx / (NHASH * Sz);
  int rem = idx % (NHASH * Sz);
  int h = rem / Sz;
  int s = rem % Sz;
  int bb = bh / Hh, hd = bh % Hh;
  const float* q = QK + ((size_t)(bb * Sz + s)) * Dz + hd * DHz;

  float r[NBz / 2];
#pragma unroll
  for (int i = 0; i < NBz / 2; ++i) r[i] = 0.0f;
  for (int dh = 0; dh < DHz; ++dh) {
    float qv = q[dh];
    const float* rp = rot + ((size_t)dh * NHASH + h) * (NBz / 2);
#pragma unroll
    for (int i = 0; i < NBz / 2; ++i) r[i] += qv * rp[i];
  }
  int best = 0;
  float bv = r[0];
  for (int i = 1; i < NBz; ++i) {
    float v = (i < NBz / 2) ? r[i] : -r[i - NBz / 2];
    if (v > bv) { bv = v; best = i; }
  }
  buckets[idx] = best;
}

// ============================================================
// Stable counting sort per (bh, hash): 64 buckets of 4096
// ============================================================
__global__ void sort_kernel(const int* __restrict__ buckets,
                            int* __restrict__ st) {
  __shared__ int cnt[NBz];
  __shared__ int off[NBz];
  int row = blockIdx.x;   // bh*NHASH + h
  const int* brow = buckets + (size_t)row * Sz;
  int* srow = st + (size_t)row * Sz;
  if (threadIdx.x < NBz) cnt[threadIdx.x] = 0;
  __syncthreads();
  for (int s = threadIdx.x; s < Sz; s += blockDim.x)
    atomicAdd(&cnt[brow[s]], 1);
  __syncthreads();
  if (threadIdx.x == 0) {
    int acc = 0;
    for (int i = 0; i < NBz; ++i) { off[i] = acc; acc += cnt[i]; }
    for (int s = 0; s < Sz; ++s) {       // stable: token order
      int bkt = brow[s];
      srow[off[bkt]++] = s;
    }
  }
}

// ============================================================
// WMMA chunked attention, flash-style over 2 KV halves.
// one block = 64 threads (2 wave32) per (bh, chunk).
// ============================================================
__global__ void attn_kernel(const float* __restrict__ QK,
                            const float* __restrict__ V,
                            const int* __restrict__ st,
                            const int* __restrict__ src,
                            _Float16* __restrict__ O,
                            float* __restrict__ logits) {
  __shared__ _Float16 Kn[2 * BUCKET][DHz];   // 16 KB normalized keys
  __shared__ _Float16 VT[DHz][2 * BUCKET];   // 16 KB V transposed
  __shared__ float    Sc[BUCKET][BUCKET];    // 16 KB score half
  __shared__ _Float16 Ph[BUCKET][BUCKET];    //  8 KB exp(scores) half
  __shared__ int   tq[BUCKET];
  __shared__ int   tkv[2 * BUCKET];
  __shared__ int   pq[BUCKET];
  __shared__ int   pkv[2 * BUCKET];
  __shared__ float mrow[BUCKET];
  __shared__ float srow[BUCKET];
  __shared__ float frow[BUCKET];

  int bh = blockIdx.x / NCHUNK;
  int c  = blockIdx.x % NCHUNK;
  int bb = bh / Hh, hd = bh % Hh;
  int h  = c / NBz;
  int cl = c % NBz;
  int cp = (c + NCHUNK - 1) % NCHUNK;   // global look-one-back (wraps)
  int hp = cp / NBz, clp = cp % NBz;
  int tid  = threadIdx.x;               // 0..63
  int wave = tid >> 5;
  int lane = tid & 31;
  int lm = lane & 15;
  bool hi = lane >= 16;
  int aB = hi ? 8 : 0;
  int bB = hi ? 8 : 0;

  // tokens / pad flags
  {
    int t = st[((size_t)bh * NHASH + h) * Sz + cl * BUCKET + tid];
    tq[tid] = t;
    pq[tid] = (src[bb * Sz + t] == 0);
    tkv[tid] = t;
    pkv[tid] = pq[tid];
    int t2 = st[((size_t)bh * NHASH + hp) * Sz + clp * BUCKET + tid];
    tkv[tid + BUCKET] = t2;
    pkv[tid + BUCKET] = (src[bb * Sz + t2] == 0);
    mrow[tid] = -3.0e38f;
    srow[tid] = 0.0f;
  }
  __syncthreads();
  // normalized keys (rows tid, tid+64)
  for (int jj = 0; jj < 2; ++jj) {
    int j = tid + jj * BUCKET;
    const float* kq = QK + ((size_t)(bb * Sz + tkv[j])) * Dz + hd * DHz;
    float nrm = 0.0f;
#pragma unroll 8
    for (int d = 0; d < DHz; ++d) { float x = kq[d]; nrm += x * x; }
    float scale = 1.0f / fmaxf(sqrtf(nrm), 1e-12f);
#pragma unroll 8
    for (int d = 0; d < DHz; ++d) Kn[j][d] = (_Float16)(kq[d] * scale);
  }
  // V transposed: thread tid = dh column
  {
    int dh = tid;
    for (int j = 0; j < 2 * BUCKET; ++j)
      VT[dh][j] = (_Float16)V[((size_t)(bb * Sz + tkv[j])) * Dz + hd * DHz + dh];
  }
  __syncthreads();

  v8f acc[4][2] = {};   // [mtile][local ntile]; wave handles ntiles {2w,2w+1}

  for (int hv = 0; hv < 2; ++hv) {
    // ---- scores for this KV half ----
#pragma unroll
    for (int mt = 0; mt < 4; ++mt) {
      int row = mt * 16 + lm;
      const float* qp = QK + ((size_t)(bb * Sz + tq[row])) * Dz + hd * DHz;
      v16h aF[2];
#pragma unroll
      for (int ks = 0; ks < 2; ++ks) {
#pragma unroll
        for (int j = 0; j < 8; ++j) {
          aF[ks][j]     = (_Float16)qp[ks * 32 + aB + j];
          aF[ks][j + 8] = (_Float16)qp[ks * 32 + aB + 16 + j];
        }
      }
#pragma unroll
      for (int ntl = 0; ntl < 2; ++ntl) {
        int nt = wave * 2 + ntl;
        v8f s = {};
#pragma unroll
        for (int ks = 0; ks < 2; ++ks) {
          int jcol = hv * 64 + nt * 16 + lm;
          const _Float16* kp = &Kn[jcol][ks * 32 + bB];
          v16h bF = cat16(*(const v8h*)kp, *(const v8h*)(kp + 16));
          s = __builtin_amdgcn_wmma_f32_16x16x32_f16(
              false, aF[ks], false, bF, (short)0, s, false, false);
        }
#pragma unroll
        for (int r = 0; r < 8; ++r) {
          int rr = mt * 16 + r + (hi ? 8 : 0);
          int jj = nt * 16 + lm;
          int jg = hv * 64 + jj;
          float d = s[r] * 0.125f;          // DH^-0.5
          if (pq[rr] || pkv[jg]) d = MASK_VAL;
          if (tq[rr] == tkv[jg]) d = SELF_MASK;  // overrides pad mask
          Sc[rr][jj] = d;
        }
      }
    }
    __syncthreads();
    // ---- row stats (thread i = row) ----
    {
      int i = tid;
      float m_old = mrow[i];
      float mx = m_old;
      for (int j = 0; j < BUCKET; ++j) mx = fmaxf(mx, Sc[i][j]);
      float scale = __expf(m_old - mx);
      float s = srow[i] * scale;
      for (int j = 0; j < BUCKET; ++j) {
        float e = __expf(Sc[i][j] - mx);
        Ph[i][j] = (_Float16)e;
        s += e;
      }
      mrow[i] = mx;
      srow[i] = s;
      frow[i] = scale;
    }
    __syncthreads();
    // ---- rescale existing output acc, then PV accumulate ----
#pragma unroll
    for (int mt = 0; mt < 4; ++mt) {
#pragma unroll
      for (int ntl = 0; ntl < 2; ++ntl) {
        int nt = wave * 2 + ntl;
#pragma unroll
        for (int r = 0; r < 8; ++r)
          acc[mt][ntl][r] *= frow[mt * 16 + r + (hi ? 8 : 0)];
#pragma unroll
        for (int ks = 0; ks < 2; ++ks) {
          const _Float16* pp = &Ph[mt * 16 + lm][ks * 32 + aB];
          v16h aF = cat16(*(const v8h*)pp, *(const v8h*)(pp + 16));
          const _Float16* vp = &VT[nt * 16 + lm][hv * 64 + ks * 32 + bB];
          v16h bF = cat16(*(const v8h*)vp, *(const v8h*)(vp + 16));
          acc[mt][ntl] = __builtin_amdgcn_wmma_f32_16x16x32_f16(
              false, aF, false, bF, (short)0, acc[mt][ntl], false, false);
        }
      }
    }
    __syncthreads();
  }

  // ---- write O rows (scattered by token) and per-round logits ----
#pragma unroll
  for (int mt = 0; mt < 4; ++mt)
#pragma unroll
    for (int ntl = 0; ntl < 2; ++ntl) {
      int nt = wave * 2 + ntl;
#pragma unroll
      for (int r = 0; r < 8; ++r) {
        int rr = mt * 16 + r + (hi ? 8 : 0);
        int dh = nt * 16 + lm;
        float v = acc[mt][ntl][r] / srow[rr];
        O[(((size_t)bh * NHASH + h) * Sz + tq[rr]) * DHz + dh] = (_Float16)v;
      }
    }
  logits[((size_t)bh * NHASH + h) * Sz + tq[tid]] =
      mrow[tid] + __logf(srow[tid]);
}

// ============================================================
// Combine hash rounds: softmax over per-round LSE -> f16 ATT
// ============================================================
__global__ void combine_kernel(const _Float16* __restrict__ O,
                               const float* __restrict__ logits,
                               _Float16* __restrict__ att) {
  int idx = blockIdx.x * blockDim.x + threadIdx.x;
  if (idx >= BHz * Sz) return;
  int bh = idx / Sz, s = idx % Sz;
  int bb = bh / Hh, hd = bh % Hh;
  float l[NHASH], w[NHASH];
  float mx = -3.0e38f;
#pragma unroll
  for (int r = 0; r < NHASH; ++r) {
    l[r] = logits[((size_t)bh * NHASH + r) * Sz + s];
    mx = fmaxf(mx, l[r]);
  }
  float den = 0.0f;
#pragma unroll
  for (int r = 0; r < NHASH; ++r) { w[r] = __expf(l[r] - mx); den += w[r]; }
  float invd = 1.0f / den;
  _Float16* out = att + ((size_t)(bb * Sz + s)) * Dz + hd * DHz;
  for (int dd = 0; dd < DHz; ++dd) {
    float a = 0.0f;
#pragma unroll
    for (int r = 0; r < NHASH; ++r)
      a += w[r] * (float)O[(((size_t)bh * NHASH + r) * Sz + s) * DHz + dd];
    out[dd] = (_Float16)(a * invd);
  }
}

// ============================================================
// Masked mean pool of 0.5*(x1+x2) over non-pad tokens
// ============================================================
__global__ void pool_kernel(const float* __restrict__ x1,
                            const float* __restrict__ x2,
                            const int* __restrict__ src,
                            float* __restrict__ pooled) {
  __shared__ float rs[256];
  __shared__ float rc[256];
  int d = blockIdx.x, b = blockIdx.y;
  float s_h = 0.0f, s_k = 0.0f;
  for (int s = threadIdx.x; s < Sz; s += 256) {
    if (src[b * Sz + s] != 0) {
      size_t ix = ((size_t)(b * Sz + s)) * Dz + d;
      s_h += 0.5f * (x1[ix] + x2[ix]);
      s_k += 1.0f;
    }
  }
  rs[threadIdx.x] = s_h; rc[threadIdx.x] = s_k;
  __syncthreads();
  for (int off = 128; off > 0; off >>= 1) {
    if (threadIdx.x < off) {
      rs[threadIdx.x] += rs[threadIdx.x + off];
      rc[threadIdx.x] += rc[threadIdx.x + off];
    }
    __syncthreads();
  }
  if (threadIdx.x == 0) pooled[b * Dz + d] = rs[0] / rc[0];
}

// small dense heads
__global__ void pre_kernel(const float* __restrict__ pooled,
                           const float* __restrict__ w,
                           const float* __restrict__ bias,
                           float* __restrict__ out) {
  int idx = blockIdx.x * blockDim.x + threadIdx.x;
  if (idx >= Bz * Dz) return;
  int b = idx / Dz, d = idx % Dz;
  float acc = bias[d];
  for (int k = 0; k < Dz; ++k) acc += pooled[b * Dz + k] * w[k * Dz + d];
  out[idx] = fmaxf(acc, 0.0f);
}

__global__ void cls_kernel(const float* __restrict__ pre,
                           const float* __restrict__ w,
                           const float* __restrict__ bias,
                           float* __restrict__ out) {
  int idx = blockIdx.x * blockDim.x + threadIdx.x;
  if (idx >= Bz * NCLS) return;
  int b = idx / NCLS, c = idx % NCLS;
  float acc = bias[c];
  for (int k = 0; k < Dz; ++k) acc += pre[b * Dz + k] * w[k * NCLS + c];
  out[idx] = acc;
}

// ============================================================
extern "C" void kernel_launch(void* const* d_in, const int* in_sizes, int n_in,
                              void* d_out, int out_size, void* d_ws, size_t ws_size,
                              hipStream_t stream) {
  (void)in_sizes; (void)n_in; (void)out_size; (void)ws_size;
  const int*   src   = (const int*)d_in[0];
  // d_in[1] = source_lengths (unused; pad derived from src==0)
  const float* rot   = (const float*)d_in[2];
  const float* emb   = (const float*)d_in[3];
  const float* ln1g  = (const float*)d_in[4];
  const float* ln1b  = (const float*)d_in[5];
  const float* wqk   = (const float*)d_in[6];
  const float* wv    = (const float*)d_in[7];
  const float* wo    = (const float*)d_in[8];
  const float* bo    = (const float*)d_in[9];
  const float* ln2g  = (const float*)d_in[10];
  const float* ln2b  = (const float*)d_in[11];
  const float* w1    = (const float*)d_in[12];
  const float* b1    = (const float*)d_in[13];
  const float* w2    = (const float*)d_in[14];
  const float* b2    = (const float*)d_in[15];
  const float* prew  = (const float*)d_in[16];
  const float* preb  = (const float*)d_in[17];
  const float* clsw  = (const float*)d_in[18];
  const float* clsb  = (const float*)d_in[19];
  float* out = (float*)d_out;

  // ---- workspace layout (bytes) ----
  const size_t MB = 1ull << 20;
  char* ws = (char*)d_ws;
  float*    X1   = (float*)(ws + 0 * MB);       // 32 MB
  float*    X2   = (float*)(ws + 32 * MB);      // 32 MB
  _Float16* XNh  = (_Float16*)(ws + 64 * MB);   // 16 MB  (LN output, f16)
  _Float16* ATTh = (_Float16*)(ws + 80 * MB);   // 16 MB  (combined attn, f16)
  float*    QK   = (float*)(ws + 96 * MB);      // 32 MB
  float*    Vb   = (float*)(ws + 128 * MB);     // 32 MB
  _Float16* FFHh = (_Float16*)(ws + 96 * MB);   // 64 MB  (reuses QK+V region)
  _Float16* O    = (_Float16*)(ws + 160 * MB);  // 64 MB
  float*    LGT  = (float*)(ws + 224 * MB);     // 2 MB
  int*      ST   = (int*)(ws + 226 * MB);       // 2 MB
  int*      BKT  = (int*)(ws + 228 * MB);       // 2 MB
  _Float16* WT   = (_Float16*)(ws + 230 * MB);  // 2 MB packed weight
  float*    POOL = (float*)(ws + 232 * MB);     // small
  float*    PRE  = (float*)(ws + 232 * MB + 64 * 1024);

  // 1. embedding + positional encoding
  embed_kernel<<<BSz, 256, 0, stream>>>(src, emb, X1, X2);

  for (int l = 0; l < Lz; ++l) {
    const float* wqkL = wqk + (size_t)l * Dz * Dz;
    const float* wvL  = wv  + (size_t)l * Dz * Dz;
    const float* woL  = wo  + (size_t)l * Dz * Dz;
    const float* boL  = bo  + (size_t)l * Dz;
    const float* w1L  = w1  + (size_t)l * Dz * 4 * Dz;
    const float* b1L  = b1  + (size_t)l * 4 * Dz;
    const float* w2L  = w2  + (size_t)l * 4 * Dz * Dz;
    const float* b2L  = b2  + (size_t)l * Dz;
    const float* rotL = rot + (size_t)l * DHz * NHASH * (NBz / 2);

    // ---- attention branch: x1 += attn(LN(x2)) ----
    layernorm_kernel<<<BSz, 256, 0, stream>>>(X2, ln1g + l * Dz, ln1b + l * Dz, XNh);
    packw_kernel<<<(Dz * Dz + 255) / 256, 256, 0, stream>>>(wqkL, WT, Dz, Dz);
    gemm_wmma_kernel<<<dim3(Dz / 64, BSz / 128), 128, 0, stream>>>(
        XNh, WT, nullptr, nullptr, QK, nullptr, BSz, Dz, Dz, 0);
    packw_kernel<<<(Dz * Dz + 255) / 256, 256, 0, stream>>>(wvL, WT, Dz, Dz);
    gemm_wmma_kernel<<<dim3(Dz / 64, BSz / 128), 128, 0, stream>>>(
        XNh, WT, nullptr, nullptr, Vb, nullptr, BSz, Dz, Dz, 0);

    hash_kernel<<<(BHz * NHASH * Sz + 255) / 256, 256, 0, stream>>>(QK, rotL, BKT);
    sort_kernel<<<BHz * NHASH, 256, 0, stream>>>(BKT, ST);
    attn_kernel<<<BHz * NCHUNK, 64, 0, stream>>>(QK, Vb, ST, src, O, LGT);
    combine_kernel<<<(BHz * Sz + 255) / 256, 256, 0, stream>>>(O, LGT, ATTh);

    packw_kernel<<<(Dz * Dz + 255) / 256, 256, 0, stream>>>(woL, WT, Dz, Dz);
    gemm_wmma_kernel<<<dim3(Dz / 64, BSz / 128), 128, 0, stream>>>(
        ATTh, WT, boL, X1, X1, nullptr, BSz, Dz, Dz, 0);

    // ---- FFN branch: x2 += ff(LN(x1)) ----
    layernorm_kernel<<<BSz, 256, 0, stream>>>(X1, ln2g + l * Dz, ln2b + l * Dz, XNh);
    packw_kernel<<<(Dz * 4 * Dz + 255) / 256, 256, 0, stream>>>(w1L, WT, Dz, 4 * Dz);
    gemm_wmma_kernel<<<dim3((4 * Dz) / 64, BSz / 128), 128, 0, stream>>>(
        XNh, WT, b1L, nullptr, nullptr, FFHh, BSz, 4 * Dz, Dz, 1 /*gelu*/);
    packw_kernel<<<(4 * Dz * Dz + 255) / 256, 256, 0, stream>>>(w2L, WT, 4 * Dz, Dz);
    gemm_wmma_kernel<<<dim3(Dz / 64, BSz / 128), 128, 0, stream>>>(
        FFHh, WT, b2L, X2, X2, nullptr, BSz, Dz, 4 * Dz, 0);
  }

  // 3. pool + heads
  pool_kernel<<<dim3(Dz, Bz), 256, 0, stream>>>(X1, X2, src, POOL);
  pre_kernel<<<(Bz * Dz + 255) / 256, 256, 0, stream>>>(POOL, prew, preb, PRE);
  cls_kernel<<<1, 256, 0, stream>>>(PRE, clsw, clsb, out);
}
